// ScaleDotAttention_18322330484921
// MI455X (gfx1250) — compile-verified, matrix operand order checked by reading, not run on hardware
//
#include <hip/hip_runtime.h>

// =====================================================================
// Attention block for MI455X (gfx1250, wave32, WMMA).
//   x:[8,2048,1024] f32, W_qkv:[3072,1024] f32, W_out:[1024,1024] f32
// Pipeline (bf16 WMMA 16x16x32, fp32 accumulate):
//   0) cvt      : x, W_qkv, W_out -> bf16 once (kills redundant cvt work)
//   1) qkv_proj : y = x @ Wqkv^T, interleaved split o=3h+t -> q,k (bf16
//      [M,H]) and v stored transposed (bf16 [B,H,S]).
//   2) attn     : per 16 query rows: scores (16x2048) in LDS, exact
//      softmax in LDS, then attn @ v -> ctx (bf16 [M,H]).
//   3) out_proj : out = ctx @ Wout^T -> fp32 d_out.
// Workspace: 5x 32MB (q,k,vt,ctx,xb) + 6MB (Wqkv) + 2MB (Wout) ~= 176MB.
// =====================================================================

typedef __attribute__((ext_vector_type(16))) __bf16 v16bf;
typedef __attribute__((ext_vector_type(8)))  __bf16 v8bf;
typedef __attribute__((ext_vector_type(4)))  __bf16 v4bf;
typedef __attribute__((ext_vector_type(4)))  float  v4f;
typedef __attribute__((ext_vector_type(8)))  float  v8f;

#define HDIM  1024
#define SEQ   2048
#define BATCH 8
#define MROWS (BATCH * SEQ)   // 16384
#define NQKV  (3 * HDIM)      // 3072
#define SST   2052            // LDS score row stride (fp32) -> 4-bank skew/row

__device__ __forceinline__ v8f wmma_bf16(v16bf a, v16bf b, v8f c) {
  return __builtin_amdgcn_wmma_f32_16x16x32_bf16(false, a, false, b,
                                                 (short)0, c, false, false);
}

__device__ __forceinline__ v8f vzero8() {
  return v8f{0.f, 0.f, 0.f, 0.f, 0.f, 0.f, 0.f, 0.f};
}

// A/B fragment (16x32 / 32x16 bf16) from row-major bf16 [rows, ld]:
//   lane%16 -> M (or N) row; lane/16 -> K-interleave half.
//   element e<8 : K = k0 + 8*half + e ; e>=8 : K = k0 + 16 + 8*half + (e-8)
__device__ __forceinline__ v16bf frag_bf16(const __bf16* base, int row0,
                                           int ld, int k0) {
  const int lane = threadIdx.x & 31;
  const __bf16* p = base + (size_t)(row0 + (lane & 15)) * ld + k0 +
                    ((lane >> 4) << 3);
  v8bf lo = *(const v8bf*)p;
  v8bf hi = *(const v8bf*)(p + 16);
  v16bf f;
#pragma unroll
  for (int i = 0; i < 8; ++i) { f[i] = lo[i]; f[i + 8] = hi[i]; }
  return f;
}

// Same fragment sourced from fp32 (LDS scores), converted to bf16.
__device__ __forceinline__ v16bf frag_f32(const float* base, int row0,
                                          int ld, int k0) {
  const int lane = threadIdx.x & 31;
  const float* p = base + (size_t)(row0 + (lane & 15)) * ld + k0 +
                   ((lane >> 4) << 3);
  v16bf f;
#pragma unroll
  for (int i = 0; i < 8; ++i) {
    f[i]     = (__bf16)p[i];
    f[i + 8] = (__bf16)p[i + 16];
  }
  return f;
}

// ---------------------------------------------------------------------
// Kernel 0: one-shot fp32 -> bf16 conversion (vectorized 4 elems/thread).
// ---------------------------------------------------------------------
__global__ __launch_bounds__(256) void cvt_bf16_kernel(
    const float* __restrict__ in, __bf16* __restrict__ out, int n4) {
  const int i = blockIdx.x * 256 + threadIdx.x;
  if (i < n4) {
    v4f v = *(const v4f*)(in + (size_t)i * 4);
    v4bf o;
#pragma unroll
    for (int e = 0; e < 4; ++e) o[e] = (__bf16)v[e];
    *(v4bf*)(out + (size_t)i * 4) = o;
  }
}

// ---------------------------------------------------------------------
// Kernel 1: QKV projection. 128x128 tile per block, 8 waves (2x4),
// each wave: 4 row-tiles x 2 col-tiles of 16x16. K-loop over 1024.
// ---------------------------------------------------------------------
__global__ __launch_bounds__(256) void qkv_proj_kernel(
    const __bf16* __restrict__ x, const __bf16* __restrict__ Wqkv,
    __bf16* __restrict__ q, __bf16* __restrict__ k,
    __bf16* __restrict__ vt) {
  const int lane = threadIdx.x & 31;
  const int w    = threadIdx.x >> 5;
  const int m0   = blockIdx.y * 128 + (w >> 2) * 64;
  const int n0   = blockIdx.x * 128 + (w & 3) * 32;

  v8f acc[4][2];
#pragma unroll
  for (int i = 0; i < 4; ++i) {
    acc[i][0] = vzero8();
    acc[i][1] = vzero8();
  }

  for (int kk = 0; kk < HDIM; kk += 32) {
    v16bf b0 = frag_bf16(Wqkv, n0,      HDIM, kk);
    v16bf b1 = frag_bf16(Wqkv, n0 + 16, HDIM, kk);
#pragma unroll
    for (int i = 0; i < 4; ++i) {
      v16bf a = frag_bf16(x, m0 + 16 * i, HDIM, kk);
      acc[i][0] = wmma_bf16(a, b0, acc[i][0]);
      acc[i][1] = wmma_bf16(a, b1, acc[i][1]);
    }
  }

  const int half = lane >> 4, nl = lane & 15;
#pragma unroll
  for (int i = 0; i < 4; ++i)
#pragma unroll
    for (int j = 0; j < 2; ++j)
#pragma unroll
      for (int r = 0; r < 8; ++r) {
        const int m = m0 + 16 * i + r + 8 * half;     // global row (b*S+s)
        const int o = n0 + 16 * j + nl;               // output channel
        const int h = o / 3, t = o - 3 * h;           // interleaved split
        const __bf16 val = (__bf16)acc[i][j][r];
        if (t == 0)      q[(size_t)m * HDIM + h] = val;
        else if (t == 1) k[(size_t)m * HDIM + h] = val;
        else {                                        // v stored transposed
          const int b = m >> 11, s = m & (SEQ - 1);
          vt[((size_t)b * HDIM + h) * SEQ + s] = val;
        }
      }
}

// ---------------------------------------------------------------------
// Kernel 2: attention for one 16-query-row tile per block (8 waves).
// Phase 1: scores (16x2048, scaled) -> LDS. Each wave owns 256 cols;
//          K outer / column-tile inner so the q A-fragment is loaded
//          once per K-step and reused across 16 WMMAs.
// Phase 2: exact softmax in LDS (16 threads per row).
// Phase 3: ctx = attn @ v. Each wave owns 128 H cols; A-fragment from
//          LDS reused across 8 WMMAs per K-step.
// ---------------------------------------------------------------------
__global__ __launch_bounds__(256) void attn_kernel(
    const __bf16* __restrict__ q, const __bf16* __restrict__ k,
    const __bf16* __restrict__ vt, __bf16* __restrict__ ctx) {
  extern __shared__ float smem[];
  float* sc      = smem;               // [16][SST] scores
  float* red     = smem + 16 * SST;    // [16][16] reduction scratch
  float* rowstat = red + 256;          // [16] row max / inv-sum

  const int lane = threadIdx.x & 31;
  const int w    = threadIdx.x >> 5;
  const int half = lane >> 4, nl = lane & 15;
  const int m0   = blockIdx.x * 16;    // global query row base
  const int b    = m0 >> 11;

  const __bf16* qrow  = q  + (size_t)m0 * HDIM;
  const __bf16* krow  = k  + ((size_t)b << 11) * HDIM;
  const __bf16* vbase = vt + (size_t)b * HDIM * SEQ;

  // ---- Phase 1: scores = (q @ k^T) * 1/sqrt(H) ----
  const int nc0 = w * 256;
  {
    v8f acc[16];
#pragma unroll
    for (int j = 0; j < 16; ++j) acc[j] = vzero8();

    for (int kk = 0; kk < HDIM; kk += 32) {
      v16bf a = frag_bf16(qrow, 0, HDIM, kk);
#pragma unroll
      for (int j = 0; j < 16; ++j) {
        v16bf bb = frag_bf16(krow, nc0 + 16 * j, HDIM, kk);
        acc[j] = wmma_bf16(a, bb, acc[j]);
      }
    }
#pragma unroll
    for (int j = 0; j < 16; ++j)
#pragma unroll
      for (int r = 0; r < 8; ++r)
        sc[(size_t)(r + 8 * half) * SST + nc0 + 16 * j + nl] =
            acc[j][r] * 0.03125f;  // 1/sqrt(1024)
  }
  __syncthreads();

  // ---- Phase 2: softmax, 16 threads per row ----
  const int row = threadIdx.x >> 4;
  const int t   = threadIdx.x & 15;
  float mx = -3.0e38f;
  for (int c = t; c < SEQ; c += 16) mx = fmaxf(mx, sc[row * SST + c]);
  red[row * 16 + t] = mx;
  __syncthreads();
  if (t == 0) {
    float m2 = red[row * 16];
    for (int i = 1; i < 16; ++i) m2 = fmaxf(m2, red[row * 16 + i]);
    rowstat[row] = m2;
  }
  __syncthreads();
  const float rm = rowstat[row];
  float sum = 0.f;
  for (int c = t; c < SEQ; c += 16) {
    float e = __expf(sc[row * SST + c] - rm);
    sc[row * SST + c] = e;
    sum += e;
  }
  red[row * 16 + t] = sum;
  __syncthreads();
  if (t == 0) {
    float s2 = 0.f;
    for (int i = 0; i < 16; ++i) s2 += red[row * 16 + i];
    rowstat[row] = 1.0f / s2;
  }
  __syncthreads();
  const float inv = rowstat[row];
  for (int c = t; c < SEQ; c += 16) sc[row * SST + c] *= inv;
  __syncthreads();

  // ---- Phase 3: ctx = attn @ v (v^T rows give [N,K] layout) ----
  const int h0 = w * 128;
  {
    v8f acc[8];
#pragma unroll
    for (int j = 0; j < 8; ++j) acc[j] = vzero8();

    for (int kk = 0; kk < SEQ; kk += 32) {
      v16bf a = frag_f32(sc, 0, SST, kk);                 // LDS, f32->bf16
#pragma unroll
      for (int j = 0; j < 8; ++j) {
        v16bf bb = frag_bf16(vbase, h0 + 16 * j, SEQ, kk);
        acc[j] = wmma_bf16(a, bb, acc[j]);
      }
    }
#pragma unroll
    for (int j = 0; j < 8; ++j)
#pragma unroll
      for (int r = 0; r < 8; ++r) {
        const int m = m0 + r + 8 * half;
        const int h = h0 + 16 * j + nl;
        ctx[(size_t)m * HDIM + h] = (__bf16)acc[j][r];
      }
  }
}

// ---------------------------------------------------------------------
// Kernel 3: output projection, same tiling as kernel 1, fp32 output.
// ---------------------------------------------------------------------
__global__ __launch_bounds__(256) void out_proj_kernel(
    const __bf16* __restrict__ ctx, const __bf16* __restrict__ Wout,
    float* __restrict__ out) {
  const int lane = threadIdx.x & 31;
  const int w    = threadIdx.x >> 5;
  const int m0   = blockIdx.y * 128 + (w >> 2) * 64;
  const int n0   = blockIdx.x * 128 + (w & 3) * 32;

  v8f acc[4][2];
#pragma unroll
  for (int i = 0; i < 4; ++i) {
    acc[i][0] = vzero8();
    acc[i][1] = vzero8();
  }

  for (int kk = 0; kk < HDIM; kk += 32) {
    v16bf b0 = frag_bf16(Wout, n0,      HDIM, kk);
    v16bf b1 = frag_bf16(Wout, n0 + 16, HDIM, kk);
#pragma unroll
    for (int i = 0; i < 4; ++i) {
      v16bf a = frag_bf16(ctx, m0 + 16 * i, HDIM, kk);
      acc[i][0] = wmma_bf16(a, b0, acc[i][0]);
      acc[i][1] = wmma_bf16(a, b1, acc[i][1]);
    }
  }

  const int half = lane >> 4, nl = lane & 15;
#pragma unroll
  for (int i = 0; i < 4; ++i)
#pragma unroll
    for (int j = 0; j < 2; ++j)
#pragma unroll
      for (int r = 0; r < 8; ++r)
        out[(size_t)(m0 + 16 * i + r + 8 * half) * HDIM + n0 + 16 * j + nl] =
            acc[i][j][r];
}

// ---------------------------------------------------------------------
extern "C" void kernel_launch(void* const* d_in, const int* in_sizes, int n_in,
                              void* d_out, int out_size, void* d_ws,
                              size_t ws_size, hipStream_t stream) {
  const float* x    = (const float*)d_in[0];
  const float* Wqkv = (const float*)d_in[1];
  const float* Wout = (const float*)d_in[2];
  float* out = (float*)d_out;

  const size_t elems = (size_t)MROWS * HDIM;  // 16.78M elements
  __bf16* qb    = (__bf16*)d_ws;
  __bf16* kb    = qb    + elems;
  __bf16* vtb   = kb    + elems;              // [B][H][S]
  __bf16* ctx   = vtb   + elems;
  __bf16* xb    = ctx   + elems;
  __bf16* wqkvb = xb    + elems;              // [3072][1024]
  __bf16* woutb = wqkvb + (size_t)NQKV * HDIM;

  // 0) one-shot fp32 -> bf16 conversions
  {
    const int nx = (int)(elems / 4);
    cvt_bf16_kernel<<<(nx + 255) / 256, 256, 0, stream>>>(x, xb, nx);
    const int nq = NQKV * HDIM / 4;
    cvt_bf16_kernel<<<(nq + 255) / 256, 256, 0, stream>>>(Wqkv, wqkvb, nq);
    const int no = HDIM * HDIM / 4;
    cvt_bf16_kernel<<<(no + 255) / 256, 256, 0, stream>>>(Wout, woutb, no);
  }

  // 1) QKV projection
  qkv_proj_kernel<<<dim3(NQKV / 128, MROWS / 128), 256, 0, stream>>>(
      xb, wqkvb, qb, kb, vtb);

  // 2) attention
  const size_t lds_bytes = (16 * SST + 256 + 16) * sizeof(float);  // ~132 KB
  (void)hipFuncSetAttribute((const void*)attn_kernel,
                            hipFuncAttributeMaxDynamicSharedMemorySize,
                            (int)lds_bytes);
  attn_kernel<<<MROWS / 16, 256, lds_bytes, stream>>>(qb, kb, vtb, ctx);

  // 3) output projection
  out_proj_kernel<<<dim3(HDIM / 128, MROWS / 128), 256, 0, stream>>>(
      ctx, woutb, out);
}
